// QFFN_24129126269084
// MI455X (gfx1250) — 8-bit matrix operand order repaired and checked by reading, compiled, not run on hardware
//
#include <hip/hip_runtime.h>
#include <hip/hip_bf16.h>
#include <cstdint>

#define BATCH 16
#define HH 48
#define WW 48
#define DIMC 256
#define HDIMC 512
#define MTOT (BATCH * HH * WW)

typedef int v8i __attribute__((ext_vector_type(8)));
typedef int qv4i __attribute__((vector_size(16)));

#if defined(__has_builtin)
#if __has_builtin(__builtin_amdgcn_global_load_async_to_lds_b128) && \
    __has_builtin(__builtin_amdgcn_s_wait_asynccnt)
#define QFFN_ASYNC_LDS 1
#endif
#if __has_builtin(__builtin_amdgcn_sdot4)
#define QFFN_SDOT4 1
#endif
#endif
#ifndef QFFN_ASYNC_LDS
#define QFFN_ASYNC_LDS 0
#endif
#ifndef QFFN_SDOT4
#define QFFN_SDOT4 0
#endif


__global__ __launch_bounds__(32) void qffn_init_slots_k(float* slots) {
    if (threadIdx.x < 16) slots[threadIdx.x] = 0.0f;
}

__global__ __launch_bounds__(256) void qffn_absmax_k(const float* __restrict__ x,
                                                     size_t n,
                                                     float* __restrict__ slot) {
    __shared__ float red[256];
    float mx = 0.0f;
    for (size_t i = (size_t)blockIdx.x * 256 + threadIdx.x; i < n;
         i += (size_t)gridDim.x * 256)
        mx = fmaxf(mx, fabsf(x[i]));
    red[threadIdx.x] = mx;
    __syncthreads();
    for (int s = 128; s > 0; s >>= 1) {
        if ((int)threadIdx.x < s)
            red[threadIdx.x] = fmaxf(red[threadIdx.x], red[threadIdx.x + s]);
        __syncthreads();
    }
    if (threadIdx.x == 0)
        atomicMax((unsigned int*)slot, __float_as_uint(red[0]));
}

__device__ __forceinline__ float qffn_slot_scale(const float* slot) {
    return fmaxf(slot[0], 1e-8f) * (1.0f / 127.0f);
}

__global__ __launch_bounds__(256) void qffn_quant_k(const float* __restrict__ x,
                                                    const float* __restrict__ slot,
                                                    int8_t* __restrict__ q, size_t n) {
    size_t i = (size_t)blockIdx.x * 256 + threadIdx.x;
    if (i >= n) return;
    float s = qffn_slot_scale(slot);
    float r = rintf(x[i] / s);
    r = fminf(fmaxf(r, -128.0f), 127.0f);
    q[i] = (int8_t)r;
}

__global__ __launch_bounds__(256) void qffn_quant_nchw_nhwc_k(
    const float* __restrict__ x, const float* __restrict__ slot,
    int8_t* __restrict__ q, int C) {
    size_t i = (size_t)blockIdx.x * 256 + threadIdx.x;
    size_t total = (size_t)MTOT * C;
    if (i >= total) return;
    int c = (int)(i % C);
    size_t m = i / C;
    int w = (int)(m % WW);
    int h = (int)((m / WW) % HH);
    int b = (int)(m / ((size_t)WW * HH));
    float v = x[(((size_t)b * C + c) * HH + h) * WW + w];
    float s = qffn_slot_scale(slot);
    float r = rintf(v / s);
    r = fminf(fmaxf(r, -128.0f), 127.0f);
    q[i] = (int8_t)r;
}

__global__ __launch_bounds__(256) void qffn_fakequant_inplace_k(
    float* __restrict__ x, const float* __restrict__ slot, size_t n) {
    size_t i = (size_t)blockIdx.x * 256 + threadIdx.x;
    if (i >= n) return;
    float s = qffn_slot_scale(slot);
    float r = rintf(x[i] / s);
    r = fminf(fmaxf(r, -128.0f), 127.0f);
    x[i] = r * s;
}


__global__ __launch_bounds__(256) void qffn_dwconv_k(
    const int8_t* __restrict__ qin, const int8_t* __restrict__ qw,
    const float* __restrict__ sA, const float* __restrict__ sW,
    const float* __restrict__ bias, float* __restrict__ out, int C,
    int gelu_flag) {
    size_t i = (size_t)blockIdx.x * 256 + threadIdx.x;
    size_t total = (size_t)MTOT * C;
    if (i >= total) return;
    int c = (int)(i % C);
    size_t m = i / C;
    int w = (int)(m % WW);
    int h = (int)((m / WW) % HH);
    int b = (int)(m / ((size_t)WW * HH));
    int g4 = (c >> 2) << 2;
    int acc = 0;
#pragma unroll
    for (int dy = 0; dy < 3; ++dy) {
        int h2 = h + dy - 1;
        if (h2 < 0 || h2 >= HH) continue;
#pragma unroll
        for (int dx = 0; dx < 3; ++dx) {
            int w2 = w + dx - 1;
            if (w2 < 0 || w2 >= WW) continue;
            const int8_t* ip = qin + ((size_t)(b * HH + h2) * WW + w2) * C + g4;
            const int8_t* wp = qw + (size_t)c * 36 + dy * 3 + dx;
#if QFFN_SDOT4
            int av = *(const int*)ip;
            int wv = (int)(unsigned char)wp[0] | ((int)(unsigned char)wp[9] << 8) |
                     ((int)(unsigned char)wp[18] << 16) |
                     ((int)(unsigned char)wp[27] << 24);
            acc = __builtin_amdgcn_sdot4(av, wv, acc, false);
#else
#pragma unroll
            for (int ci = 0; ci < 4; ++ci)
                acc += (int)ip[ci] * (int)wp[ci * 9];
#endif
        }
    }
    float scale = qffn_slot_scale(sA) * qffn_slot_scale(sW);
    float v = (float)acc * scale + bias[c];
    if (gelu_flag) v = 0.5f * v * (1.0f + erff(v * 0.70710678118654752f));
    out[i] = v;
}


template <int K, int GELU>
__global__ __launch_bounds__(256) void qffn_gemm_iu8_k(
    const int8_t* __restrict__ Aq, const int8_t* __restrict__ Wq,
    const float* __restrict__ sA, const float* __restrict__ sW,
    const float* __restrict__ bias, float* __restrict__ out, int N) {
    __shared__ alignas(16) int8_t lds_w[128 * K];

    const int n0 = blockIdx.x * 128;
    const int m_blk = blockIdx.y * 128;

#if QFFN_ASYNC_LDS
    {
        typedef __attribute__((address_space(1))) qv4i glb_v4i;
        typedef __attribute__((address_space(3))) qv4i lds_v4i;
        glb_v4i* src = (glb_v4i*)(Wq + (size_t)n0 * K);
        lds_v4i* dst = (lds_v4i*)lds_w;
        for (int i = threadIdx.x; i < (128 * K) >> 4; i += 256)
            __builtin_amdgcn_global_load_async_to_lds_b128(src + i, dst + i, 0, 0);
        __builtin_amdgcn_s_wait_asynccnt(0);
    }
#else
    {
        const int4* src = (const int4*)(Wq + (size_t)n0 * K);
        int4* dst = (int4*)lds_w;
        for (int i = threadIdx.x; i < (128 * K) >> 4; i += 256) dst[i] = src[i];
    }
#endif
    __syncthreads();

    const int wave = threadIdx.x >> 5;
    const int lane = threadIdx.x & 31;
    const int hl = lane >> 4;
    const int l16 = lane & 15;
    const size_t arow = (size_t)(m_blk + wave * 16 + l16) * K;

    v8i acc[8];
#pragma unroll
    for (int t = 0; t < 8; ++t) acc[t] = (v8i){0, 0, 0, 0, 0, 0, 0, 0};

    for (int kb = 0; kb < K; kb += 64) {
        const int8_t* ap = Aq + arow + kb + hl * 32;
        __builtin_prefetch(ap + 64, 0, 1);
        int2 t0 = *(const int2*)(ap + 0);
        int2 t1 = *(const int2*)(ap + 8);
        int2 t2 = *(const int2*)(ap + 16);
        int2 t3 = *(const int2*)(ap + 24);
        v8i a;
        a[0] = t0.x; a[1] = t0.y; a[2] = t1.x; a[3] = t1.y;
        a[4] = t2.x; a[5] = t2.y; a[6] = t3.x; a[7] = t3.y;
#pragma unroll
        for (int nt = 0; nt < 8; ++nt) {
            const int8_t* bp =
                lds_w + (size_t)(nt * 16 + l16) * K + kb + hl * 32;
            int4 u0 = *(const int4*)(bp + 0);
            int4 u1 = *(const int4*)(bp + 16);
            v8i bfr;
            bfr[0] = u0.x; bfr[1] = u0.y; bfr[2] = u0.z; bfr[3] = u0.w;
            bfr[4] = u1.x; bfr[5] = u1.y; bfr[6] = u1.z; bfr[7] = u1.w;
            acc[nt] = __builtin_amdgcn_wmma_i32_16x16x64_iu8(
                true, a, true, bfr, acc[nt], false, false);
        }
    }

    const float scale = qffn_slot_scale(sA) * qffn_slot_scale(sW);
#pragma unroll
    for (int nt = 0; nt < 8; ++nt) {
        const int n = n0 + nt * 16 + l16;
        const float bn = bias[n];
#pragma unroll
        for (int j = 0; j < 8; ++j) {
            int m = m_blk + wave * 16 + hl * 8 + j;
            float v = (float)acc[nt][j] * scale + bn;
            if (GELU) v = 0.5f * v * (1.0f + erff(v * 0.70710678118654752f));
            out[(size_t)m * N + n] = v;
        }
    }
}


__global__ __launch_bounds__(256) void qffn_ln_k(const float* __restrict__ in,
                                                 const float* __restrict__ g,
                                                 const float* __restrict__ bb,
                                                 float* __restrict__ out) {
    __shared__ float s1[256];
    __shared__ float s2[256];
    int m = blockIdx.x;
    int c = threadIdx.x;
    float v = in[(size_t)m * DIMC + c];
    s1[c] = v;
    s2[c] = v * v;
    __syncthreads();
    for (int s = 128; s > 0; s >>= 1) {
        if (c < s) {
            s1[c] += s1[c + s];
            s2[c] += s2[c + s];
        }
        __syncthreads();
    }
    float mu = s1[0] * (1.0f / DIMC);
    float var = s2[0] * (1.0f / DIMC) - mu * mu;
    float y = (v - mu) * rsqrtf(var + 1e-5f) * g[c] + bb[c];
    int w = m % WW;
    int h = (m / WW) % HH;
    int b = m / (WW * HH);
    out[(((size_t)b * DIMC + c) * HH + h) * WW + w] = y;
}


extern "C" void kernel_launch(void* const* d_in, const int* in_sizes, int n_in,
                              void* d_out, int out_size, void* d_ws,
                              size_t ws_size, hipStream_t stream) {
    (void)in_sizes; (void)n_in; (void)out_size; (void)ws_size;

    const float* x     = (const float*)d_in[0];
    const float* w_dw1 = (const float*)d_in[1];
    const float* b_dw1 = (const float*)d_in[2];
    const float* w_pw1 = (const float*)d_in[3];
    const float* b_pw1 = (const float*)d_in[4];
    const float* w_dw2 = (const float*)d_in[5];
    const float* b_dw2 = (const float*)d_in[6];
    const float* w_pw2 = (const float*)d_in[7];
    const float* b_pw2 = (const float*)d_in[8];
    const float* w_pw3 = (const float*)d_in[9];
    const float* b_pw3 = (const float*)d_in[10];
    const float* w_dw3 = (const float*)d_in[11];
    const float* b_dw3 = (const float*)d_in[12];
    const float* ng    = (const float*)d_in[13];
    const float* nb    = (const float*)d_in[14];
    float* out = (float*)d_out;

    uint8_t* ws = (uint8_t*)d_ws;
    size_t off = 0;
    auto alloc = [&](size_t bytes) -> void* {
        off = (off + 255) & ~(size_t)255;
        void* p = ws + off;
        off += bytes;
        return p;
    };

    float* slots  = (float*)alloc(16 * sizeof(float));
    int8_t* q_dw1 = (int8_t*)alloc((size_t)DIMC * 36);
    int8_t* q_pw1 = (int8_t*)alloc((size_t)HDIMC * DIMC);
    int8_t* q_dw2 = (int8_t*)alloc((size_t)HDIMC * 36);
    int8_t* q_pw2 = (int8_t*)alloc((size_t)HDIMC * HDIMC);
    int8_t* q_pw3 = (int8_t*)alloc((size_t)DIMC * HDIMC);
    int8_t* q_dw3 = (int8_t*)alloc((size_t)DIMC * 36);
    int8_t* q0    = (int8_t*)alloc((size_t)MTOT * HDIMC);
    int8_t* q1    = (int8_t*)alloc((size_t)MTOT * HDIMC);
    float* fbuf   = (float*)alloc((size_t)MTOT * HDIMC * sizeof(float));

    qffn_init_slots_k<<<1, 32, 0, stream>>>(slots);

    auto amax = [&](const float* p, size_t n, int slot) {
        int blocks = (int)((n + 255) / 256);
        if (blocks > 2048) blocks = 2048;
        qffn_absmax_k<<<blocks, 256, 0, stream>>>(p, n, slots + slot);
    };
    auto quant = [&](const float* p, int slot, int8_t* q, size_t n) {
        qffn_quant_k<<<(int)((n + 255) / 256), 256, 0, stream>>>(p, slots + slot,
                                                                 q, n);
    };

    amax(x, (size_t)BATCH * DIMC * HH * WW, 0);
    amax(w_dw1, (size_t)DIMC * 36, 1);
    amax(w_pw1, (size_t)HDIMC * DIMC, 2);
    amax(w_dw2, (size_t)HDIMC * 36, 3);
    amax(w_pw2, (size_t)HDIMC * HDIMC, 4);
    amax(w_pw3, (size_t)DIMC * HDIMC, 5);
    amax(w_dw3, (size_t)DIMC * 36, 6);

    quant(w_dw1, 1, q_dw1, (size_t)DIMC * 36);
    quant(w_pw1, 2, q_pw1, (size_t)HDIMC * DIMC);
    quant(w_dw2, 3, q_dw2, (size_t)HDIMC * 36);
    quant(w_pw2, 4, q_pw2, (size_t)HDIMC * HDIMC);
    quant(w_pw3, 5, q_pw3, (size_t)DIMC * HDIMC);
    quant(w_dw3, 6, q_dw3, (size_t)DIMC * 36);

    {
        size_t n = (size_t)MTOT * DIMC;
        qffn_quant_nchw_nhwc_k<<<(int)((n + 255) / 256), 256, 0, stream>>>(
            x, slots + 0, q0, DIMC);
    }

    {
        size_t n = (size_t)MTOT * DIMC;
        qffn_dwconv_k<<<(int)((n + 255) / 256), 256, 0, stream>>>(
            q0, q_dw1, slots + 0, slots + 1, b_dw1, fbuf, DIMC, 0);
        amax(fbuf, n, 7);
        quant(fbuf, 7, q1, n);
        dim3 g(HDIMC / 128, MTOT / 128);
        qffn_gemm_iu8_k<DIMC, 1><<<g, 256, 0, stream>>>(
            q1, q_pw1, slots + 7, slots + 2, b_pw1, fbuf, HDIMC);
        size_t n2 = (size_t)MTOT * HDIMC;
        amax(fbuf, n2, 8);
        quant(fbuf, 8, q0, n2);
    }

    {
        size_t n = (size_t)MTOT * HDIMC;
        qffn_dwconv_k<<<(int)((n + 255) / 256), 256, 0, stream>>>(
            q0, q_dw2, slots + 8, slots + 3, b_dw2, fbuf, HDIMC, 0);
        amax(fbuf, n, 9);
        quant(fbuf, 9, q1, n);
        dim3 g(HDIMC / 128, MTOT / 128);
        qffn_gemm_iu8_k<HDIMC, 1><<<g, 256, 0, stream>>>(
            q1, q_pw2, slots + 9, slots + 4, b_pw2, fbuf, HDIMC);
        amax(fbuf, n, 10);
        quant(fbuf, 10, q0, n);
    }

    {
        dim3 g(DIMC / 128, MTOT / 128);
        qffn_gemm_iu8_k<HDIMC, 0><<<g, 256, 0, stream>>>(
            q0, q_pw3, slots + 10, slots + 5, b_pw3, fbuf, DIMC);
        size_t n = (size_t)MTOT * DIMC;
        amax(fbuf, n, 11);
        quant(fbuf, 11, q1, n);
        qffn_dwconv_k<<<(int)((n + 255) / 256), 256, 0, stream>>>(
            q1, q_dw3, slots + 11, slots + 6, b_dw3, fbuf, DIMC, 1);
    }

    qffn_ln_k<<<MTOT, 256, 0, stream>>>(fbuf, ng, nb, out);
    {
        size_t n = (size_t)MTOT * DIMC;
        amax(out, n, 12);
        qffn_fakequant_inplace_k<<<(int)((n + 255) / 256), 256, 0, stream>>>(
            out, slots + 12, n);
    }
}
